// MultiPoleScoreNet_4509715661581
// MI455X (gfx1250) — matrix operand order repaired and checked by reading, not compiled, not run on hardware
//
#include <hip/hip_runtime.h>
#include <stdint.h>

typedef __attribute__((ext_vector_type(16))) _Float16 v16h;
typedef __attribute__((ext_vector_type(8)))  float    v8f;

#define C_LATENT 128
#define C_HIDDEN 64

static __device__ __forceinline__ v8f wmma_f16(v16h a, v16h b, v8f c) {
  // D = A(16x32 f16) x B(32x16 f16) + C(16x16 f32)
  return __builtin_amdgcn_wmma_f32_16x16x32_f16(false, a, false, b, (short)0, c, false, false);
}

// jax.nn.gelu (tanh approximation), branch-free.
// CDNA5 has a hardware V_TANH_F32 (TRANS op) -- use it when the builtin exists.
static __device__ __forceinline__ float gelu_tanh(float x) {
  float u = 0.7978845608028654f * (x + 0.044715f * x * x * x);
#if __has_builtin(__builtin_amdgcn_tanhf)
  float t = __builtin_amdgcn_tanhf(u);
#else
  float e = __expf(2.0f * u);          // v_exp_f32, branch-free
  float t = 1.0f - 2.0f / (e + 1.0f);
#endif
  return 0.5f * x * (1.0f + t);
}

// ---------------------------------------------------------------------------
// Pack row-major f32 weight (Ksrc x Nsrc) into WMMA B-operand layout, zero
// padded to (32*kkn x Npad).  Per k-step kk and 16-col tile t:
//   lane L holds col n = 16t + (L%16); element e holds K = 32kk + 8*(L/16) + (e<8 ? e : e+8)
// (CDNA5 ISA B layout, dense 32x16 f16 operand)
// ---------------------------------------------------------------------------
__global__ void k_prepB(const float* __restrict__ w, int Ksrc, int Npad, int Nsrc,
                        _Float16* __restrict__ out, int total) {
  int idx = blockIdx.x * blockDim.x + threadIdx.x;
  if (idx >= total) return;
  int e     = idx & 15;
  int lane  = (idx >> 4) & 31;
  int rest  = idx >> 9;
  int tiles = Npad >> 4;
  int t  = rest % tiles;
  int kk = rest / tiles;
  int k = kk * 32 + ((lane >> 4) << 3) + (e < 8 ? e : e + 8);
  int n = t * 16 + (lane & 15);
  float v = (k < Ksrc && n < Nsrc) ? w[k * Nsrc + n] : 0.f;
  out[idx] = (_Float16)v;
}

// ------------------------------ elementwise --------------------------------
__global__ void k_zero(float* p, int n) {
  int i = blockIdx.x * blockDim.x + threadIdx.x;
  if (i < n) p[i] = 0.f;
}
__global__ void k_count(const int* __restrict__ dst, int E, float* cnt) {
  int i = blockIdx.x * blockDim.x + threadIdx.x;
  if (i < E) atomicAdd(&cnt[dst[i]], 1.0f);
}
__global__ void k_recip(float* c, int n) {
  int i = blockIdx.x * blockDim.x + threadIdx.x;
  if (i < n) c[i] = 1.0f / fmaxf(c[i], 1.0f);
}
__global__ void k_scale_rows(float* a, const float* __restrict__ inv, int n) {
  int i = blockIdx.x * blockDim.x + threadIdx.x;
  if (i < n) a[i] *= inv[i >> 7];            // 128 cols per row
}
__global__ void k_add_relu(float* __restrict__ o, const float* __restrict__ a,
                           const float* __restrict__ b, int n, int relu) {
  int i = blockIdx.x * blockDim.x + threadIdx.x;
  if (i >= n) return;
  float v = a[i] + (b ? b[i] : 0.f);
  o[i] = relu ? fmaxf(v, 0.f) : v;
}

// ---------------------------------------------------------------------------
// Dense GEMM:  C[M x store_n] = A[M x Ksrc](f32) * B(prepacked f16) + bias
//              (+ add1)  (+ relu).  One wave computes a 16-row x N tile.
// A-operand layout (16x32 f16): lane L row m=L%16; element e holds
//   K = 32kk + 8*(L/16) + (e<8 ? e : e+8)
// ---------------------------------------------------------------------------
template <int TILES>
__global__ void k_gemm(const float* __restrict__ A, int M, int Ksrc,
                       const _Float16* __restrict__ B,
                       const float* __restrict__ bias, int bias_n,
                       const float* __restrict__ add1, int relu,
                       float* __restrict__ C, int ldc, int store_n) {
  int lane = threadIdx.x & 31;
  int wv   = threadIdx.x >> 5;
  int row0 = (blockIdx.x * 4 + wv) * 16;
  if (row0 >= M) return;

  int kkn  = (Ksrc + 31) >> 5;
  int rlo  = lane & 15;
  int kh   = (lane >> 4) << 3;   // 0 or 8

  v8f acc[TILES] = {};
  for (int kk = 0; kk < kkn; ++kk) {
    v16h a;
    const float* ap = A + (long)(row0 + rlo) * Ksrc + kk * 32 + kh;
#pragma unroll
    for (int i = 0; i < 16; ++i) {
      int off = (i < 8) ? i : (i + 8);
      int k = kk * 32 + kh + off;
      a[i] = (k < Ksrc) ? (_Float16)ap[off] : (_Float16)0.f;
    }
#pragma unroll
    for (int t = 0; t < TILES; ++t) {
      v16h b = *(const v16h*)(B + (((long)kk * TILES + t) * 32 + lane) * 16);
      acc[t] = wmma_f16(a, b, acc[t]);
    }
  }
  // C/D layout: lane L col n = L%16; vgpr r -> row m = r + 8*(L/16)
#pragma unroll
  for (int t = 0; t < TILES; ++t) {
    int col = t * 16 + rlo;
    if (col < store_n) {
      float bb = (bias && col < bias_n) ? bias[col] : 0.f;
#pragma unroll
      for (int r = 0; r < 8; ++r) {
        int row = row0 + kh + r;
        float v = acc[t][r] + bb;
        if (add1) v += add1[(long)row * 128 + col];
        if (relu) v = fmaxf(v, 0.f);
        C[(long)row * ldc + col] = v;
      }
    }
  }
}

// ---------------------------------------------------------------------------
// Fused edge GNO: per 16-edge tile (one wave):
//   k = gelu([yc[src], xc[dst]] @ w1 + b1) @ w2 + b2     (WMMA f16)
//   atomicAdd(out[dst], k * F[src])                      (scatter-sum)
// ---------------------------------------------------------------------------
#define EW 8  // waves (edge tiles) per block
__global__ void k_edge_gno(const int* __restrict__ esrc, const int* __restrict__ edst,
                           int E, const float* __restrict__ coords, int npts0,
                           int ylev, int xlev,
                           const float* __restrict__ F,          // y-level feats, ld 128
                           const _Float16* __restrict__ B1, const float* __restrict__ b1,
                           const _Float16* __restrict__ B2, const float* __restrict__ b2,
                           float* __restrict__ outAcc) {
  __shared__ int       s_src[EW][16];
  __shared__ int       s_dst[EW][16];
  __shared__ _Float16  s_h[EW][16][C_HIDDEN];

  int lane = threadIdx.x & 31;
  int wv   = threadIdx.x >> 5;
  long base = ((long)blockIdx.x * EW + wv) * 16;
  int  m    = lane & 15;
  int  kh   = (lane >> 4) << 3;

  // ---- build A1: 16 edges x 32 K (only K=0..3 nonzero, lanes 0-15) ----
  v16h a1 = {};
  if (lane < 16) {
    long e = base + m;
    if (e >= E) e = (long)E - 1;   // clamped dummy; store is guarded below
    int s = esrc[e], d = edst[e];
    s_src[wv][m] = s;
    s_dst[wv][m] = d;
    int si = ylev ? 2 * s : s;
    int di = xlev ? 2 * d : d;
    a1[0] = (_Float16)coords[si];
    a1[1] = (_Float16)coords[npts0 + si];
    a1[2] = (_Float16)coords[di];
    a1[3] = (_Float16)coords[npts0 + di];
    __builtin_prefetch(F + (long)s * C_LATENT, 0, 0);   // global_prefetch_b8
  }

  // ---- layer 1: h = gelu(A1 @ w1 + b1), N = 64 (4 tiles, 1 k-step) ----
  v8f h[4];
#pragma unroll
  for (int t = 0; t < 4; ++t) {
    v16h b = *(const v16h*)(B1 + ((long)t * 32 + lane) * 16);
    v8f z = {};
    h[t] = wmma_f16(a1, b, z);
  }
#pragma unroll
  for (int t = 0; t < 4; ++t) {
    int col = t * 16 + (lane & 15);
    float bb = b1[col];
#pragma unroll
    for (int r = 0; r < 8; ++r) {
      float g = gelu_tanh(h[t][r] + bb);
      s_h[wv][kh + r][col] = (_Float16)g;
    }
  }
  __syncthreads();

  // ---- layer 2: k = h @ w2 + b2, K=64 (2 k-steps), N=128 (8 tiles) ----
  v8f acc[8] = {};
#pragma unroll
  for (int kk = 0; kk < 2; ++kk) {
    v16h a2;
    const _Float16* hp = &s_h[wv][lane & 15][kk * 32 + kh];
#pragma unroll
    for (int i = 0; i < 8; ++i) a2[i] = hp[i];
#pragma unroll
    for (int i = 0; i < 8; ++i) a2[8 + i] = hp[16 + i];
#pragma unroll
    for (int t = 0; t < 8; ++t) {
      v16h b = *(const v16h*)(B2 + (((long)kk * 8 + t) * 32 + lane) * 16);
      acc[t] = wmma_f16(a2, b, acc[t]);
    }
  }

  // ---- epilogue: msg = (k + b2) * F[src], scatter-add into out[dst] ----
#pragma unroll
  for (int t = 0; t < 8; ++t) {
    int col = t * 16 + (lane & 15);
    float bb = b2[col];
#pragma unroll
    for (int r = 0; r < 8; ++r) {
      int mm = kh + r;
      long e = base + mm;
      if (e < E) {
        int s = s_src[wv][mm];
        int d = s_dst[wv][mm];
        float v = (acc[t][r] + bb) * F[(long)s * C_LATENT + col];
        atomicAdd(&outAcc[(long)d * C_LATENT + col], v);
      }
    }
  }
}

// ---------------------------------------------------------------------------
extern "C" void kernel_launch(void* const* d_in, const int* in_sizes, int n_in,
                              void* d_out, int out_size, void* d_ws, size_t ws_size,
                              hipStream_t stream) {
  const float* coords  = (const float*)d_in[0];
  const float* samples = (const float*)d_in[1];
  const float* lift_w  = (const float*)d_in[3];
  const float* lift_b  = (const float*)d_in[4];
  const float* proj_w  = (const float*)d_in[5];
  const float* proj_b  = (const float*)d_in[6];
  const float* W0_w    = (const float*)d_in[7];
  const float* W0_b    = (const float*)d_in[8];
  const float* W1_w    = (const float*)d_in[9];
  const float* W1_b    = (const float*)d_in[10];
  // g: 0=gk00, 1=gk01, 2=gk10, 3=gk11
  const float *gw1[4], *gb1[4], *gw2[4], *gb2[4];
  for (int g = 0; g < 4; ++g) {
    gw1[g] = (const float*)d_in[11 + 4 * g];
    gb1[g] = (const float*)d_in[12 + 4 * g];
    gw2[g] = (const float*)d_in[13 + 4 * g];
    gb2[g] = (const float*)d_in[14 + 4 * g];
  }
  // s: 0=e00, 1=e01, 2=e10, 3=e11
  const int *esrc[4], *edst[4];
  int E[4];
  for (int s = 0; s < 4; ++s) {
    esrc[s] = (const int*)d_in[27 + 2 * s];
    edst[s] = (const int*)d_in[28 + 2 * s];
    E[s]    = in_sizes[27 + 2 * s];
  }
  const int N0 = in_sizes[1] / 3;   // 8192
  const int N1 = (N0 + 1) / 2;      // 4096

  // ---- workspace bump allocator (256B aligned) ----
  uintptr_t cur = (uintptr_t)d_ws;
  auto alloc = [&](size_t bytes) -> void* {
    cur = (cur + 255) & ~(uintptr_t)255;
    void* p = (void*)cur;
    cur += bytes;
    return p;
  };
  float* v0     = (float*)alloc((size_t)N0 * 128 * 4);
  float* base0  = (float*)alloc((size_t)N0 * 128 * 4);
  float* vu0    = (float*)alloc((size_t)N0 * 128 * 4);
  float* accN0  = (float*)alloc((size_t)N0 * 128 * 4);
  float* d1base = (float*)alloc((size_t)N1 * 128 * 4);
  float* vd1    = (float*)alloc((size_t)N1 * 128 * 4);
  float* vu1    = (float*)alloc((size_t)N1 * 128 * 4);
  float* accN1  = (float*)alloc((size_t)N1 * 128 * 4);
  float* cnt[4];
  int cntN[4] = {N0, N1, N0, N1};   // dst level sizes of e00,e01,e10,e11
  for (int s = 0; s < 4; ++s) cnt[s] = (float*)alloc((size_t)cntN[s] * 4);

  _Float16* Blift = (_Float16*)alloc(4096 * 2);     // K 3->32, N 128
  _Float16* Bproj = (_Float16*)alloc(2048 * 2);     // K 128, N 3->16
  _Float16* BW0   = (_Float16*)alloc(16384 * 2);    // 128x128
  _Float16* BW1   = (_Float16*)alloc(16384 * 2);
  _Float16 *Bg1[4], *Bg2[4];
  for (int g = 0; g < 4; ++g) {
    Bg1[g] = (_Float16*)alloc(2048 * 2);            // K 4->32,  N 64
    Bg2[g] = (_Float16*)alloc(8192 * 2);            // K 64,     N 128
  }

  auto ew = [&](int n) { return dim3((unsigned)((n + 255) / 256)); };

  // ---- pack all weights into WMMA B-operand layout ----
  k_prepB<<<ew(4096), 256, 0, stream>>>(lift_w, 3, 128, 128, Blift, 4096);
  k_prepB<<<ew(2048), 256, 0, stream>>>(proj_w, 128, 16, 3, Bproj, 2048);
  k_prepB<<<ew(16384), 256, 0, stream>>>(W0_w, 128, 128, 128, BW0, 16384);
  k_prepB<<<ew(16384), 256, 0, stream>>>(W1_w, 128, 128, 128, BW1, 16384);
  for (int g = 0; g < 4; ++g) {
    k_prepB<<<ew(2048), 256, 0, stream>>>(gw1[g], 4, 64, 64, Bg1[g], 2048);
    k_prepB<<<ew(8192), 256, 0, stream>>>(gw2[g], 64, 128, 128, Bg2[g], 8192);
  }

  // ---- degree counts -> 1/max(cnt,1) ----
  for (int s = 0; s < 4; ++s) {
    k_zero<<<ew(cntN[s]), 256, 0, stream>>>(cnt[s], cntN[s]);
    k_count<<<ew(E[s]), 256, 0, stream>>>(edst[s], E[s], cnt[s]);
    k_recip<<<ew(cntN[s]), 256, 0, stream>>>(cnt[s], cntN[s]);
  }

  // ---- lift: v0 = samples @ lift_w + lift_b ----
  k_gemm<8><<<dim3(N0 / 64), 128, 0, stream>>>(samples, N0, 3, Blift, lift_b, 128,
                                               nullptr, 0, v0, 128, 128);

  auto egrid = [&](int e) { return dim3((unsigned)((e + 16 * EW - 1) / (16 * EW))); };

  // ---- iteration-invariant hoisting ----
  // base0 = v0 @ W0 + W0_b + gno00(v0)
  k_zero<<<ew(N0 * 128), 256, 0, stream>>>(accN0, N0 * 128);
  k_edge_gno<<<egrid(E[0]), 32 * EW, 0, stream>>>(esrc[0], edst[0], E[0], coords, N0, 0, 0,
                                                  v0, Bg1[0], gb1[0], Bg2[0], gb2[0], accN0);
  k_scale_rows<<<ew(N0 * 128), 256, 0, stream>>>(accN0, cnt[0], N0 * 128);
  k_gemm<8><<<dim3(N0 / 64), 128, 0, stream>>>(v0, N0, 128, BW0, W0_b, 128,
                                               accN0, 0, base0, 128, 128);
  // d1base = gno01(v0)
  k_zero<<<ew(N1 * 128), 256, 0, stream>>>(d1base, N1 * 128);
  k_edge_gno<<<egrid(E[1]), 32 * EW, 0, stream>>>(esrc[1], edst[1], E[1], coords, N0, 0, 1,
                                                  v0, Bg1[1], gb1[1], Bg2[1], gb2[1], d1base);
  k_scale_rows<<<ew(N1 * 128), 256, 0, stream>>>(d1base, cnt[1], N1 * 128);

  // ---- V-cycle iterations ----
  for (int it = 0; it < 3; ++it) {
    // v_down1 = relu(d1base + v_up1_prev)
    k_add_relu<<<ew(N1 * 128), 256, 0, stream>>>(vd1, d1base, it ? vu1 : nullptr,
                                                 N1 * 128, 1);
    // gno11(v_down1)
    k_zero<<<ew(N1 * 128), 256, 0, stream>>>(accN1, N1 * 128);
    k_edge_gno<<<egrid(E[3]), 32 * EW, 0, stream>>>(esrc[3], edst[3], E[3], coords, N0, 1, 1,
                                                    vd1, Bg1[3], gb1[3], Bg2[3], gb2[3], accN1);
    k_scale_rows<<<ew(N1 * 128), 256, 0, stream>>>(accN1, cnt[3], N1 * 128);
    // v_up1 = relu(v_down1 @ W1 + W1_b + gno11)
    k_gemm<8><<<dim3(N1 / 64), 128, 0, stream>>>(vd1, N1, 128, BW1, W1_b, 128,
                                                 accN1, 1, vu1, 128, 128);
    // gno10(v_up1)
    k_zero<<<ew(N0 * 128), 256, 0, stream>>>(accN0, N0 * 128);
    k_edge_gno<<<egrid(E[2]), 32 * EW, 0, stream>>>(esrc[2], edst[2], E[2], coords, N0, 1, 0,
                                                    vu1, Bg1[2], gb1[2], Bg2[2], gb2[2], accN0);
    k_scale_rows<<<ew(N0 * 128), 256, 0, stream>>>(accN0, cnt[2], N0 * 128);
    // v_up0 = relu(gno10 + base0)
    k_add_relu<<<ew(N0 * 128), 256, 0, stream>>>(vu0, accN0, base0, N0 * 128, 1);
  }

  // ---- projection: out = v_up0 @ proj_w + proj_b ----
  k_gemm<1><<<dim3(N0 / 64), 128, 0, stream>>>(vu0, N0, 128, Bproj, proj_b, 3,
                                               nullptr, 0, (float*)d_out, 3, 3);
}
